// Correlation_network_87462714016585
// MI455X (gfx1250) — matrix operand order repaired and checked by reading, not compiled, not run on hardware
//
#include <hip/hip_runtime.h>

// ---- CDNA5 (gfx1250) wave32 WMMA types ----
typedef __attribute__((ext_vector_type(16))) __bf16 v16bf;
typedef __attribute__((ext_vector_type(8)))  __bf16 v8bf;
typedef __attribute__((ext_vector_type(8)))  float  v8f;

// Problem constants (reference: B=8, H=W=48, C=64)
#define NB   8
#define HW   2304          // 48*48 rows per batch
#define KC   64            // channels (K)
#define BT   128           // output tile edge per workgroup
#define RS   72            // LDS row stride in bf16 elems (64 + 8 pad -> bank-conflict free)

__device__ __forceinline__ unsigned short f2bf_rne(float f) {
    unsigned int u = __float_as_uint(f);
    u += 0x7FFFu + ((u >> 16) & 1u);           // round-to-nearest-even to bf16
    return (unsigned short)(u >> 16);
}
__device__ __forceinline__ float bf_hi_f32(unsigned short h) {
    return __uint_as_float(((unsigned int)h) << 16);
}
__device__ __forceinline__ void split4(const float4 v, ushort4* h, ushort4* l) {
    h->x = f2bf_rne(v.x); l->x = f2bf_rne(v.x - bf_hi_f32(h->x));
    h->y = f2bf_rne(v.y); l->y = f2bf_rne(v.y - bf_hi_f32(h->y));
    h->z = f2bf_rne(v.z); l->z = f2bf_rne(v.z - bf_hi_f32(h->z));
    h->w = f2bf_rne(v.w); l->w = f2bf_rne(v.w - bf_hi_f32(h->w));
}
__device__ __forceinline__ v16bf cat8(v8bf a, v8bf b) {
    v16bf r;
#pragma unroll
    for (int i = 0; i < 8; ++i) { r[i] = a[i]; r[i + 8] = b[i]; }
    return r;
}

__global__ void __launch_bounds__(256, 2)
corr_wmma_bf16split_kernel(const float* __restrict__ A,
                           const float* __restrict__ B,
                           float* __restrict__ out)
{
    // hi/lo bf16 copies of the A-block (rows of A) and B-block (rows of B == cols of B^T)
    __shared__ __align__(16) unsigned short sAhi[BT * RS];
    __shared__ __align__(16) unsigned short sAlo[BT * RS];
    __shared__ __align__(16) unsigned short sBhi[BT * RS];
    __shared__ __align__(16) unsigned short sBlo[BT * RS];

    const int tid   = threadIdx.x;
    const int batch = blockIdx.z;
    const int tM    = blockIdx.y;
    const int tN    = blockIdx.x;

    // ---------------- load + f32 -> bf16(hi)+bf16(lo) split into LDS ----------------
    {
        const float4* Ag = (const float4*)(A + ((size_t)batch * HW + (size_t)tM * BT) * KC);
        const float4* Bg = (const float4*)(B + ((size_t)batch * HW + (size_t)tN * BT) * KC);
#pragma unroll
        for (int i = tid; i < (BT * KC) / 4; i += 256) {   // 2048 float4, 8 per thread
            const int r  = i >> 4;                          // 16 float4 per 64-elem row
            const int c  = (i & 15) << 2;
            const float4 va = Ag[i];
            const float4 vb = Bg[i];
            ushort4 h, l;
            split4(va, &h, &l);
            *(ushort4*)&sAhi[r * RS + c] = h;
            *(ushort4*)&sAlo[r * RS + c] = l;
            split4(vb, &h, &l);
            *(ushort4*)&sBhi[r * RS + c] = h;
            *(ushort4*)&sBlo[r * RS + c] = l;
        }
    }
    __syncthreads();

    // ---------------- per-wave 32x64 sub-tile: 2x4 grid of 16x16 WMMA tiles ----------------
    const int lane = tid & 31;
    const int lm   = lane & 15;
    const int g    = lane >> 4;          // lane half (ISA fragment layout selector)
    const int wave = tid >> 5;
    const int wm0  = (wave & 3) * 32;    // wave M origin inside 128x128 tile
    const int wn0  = (wave >> 2) * 64;   // wave N origin inside 128x128 tile

    v8f acc[2][4] = {};

#pragma unroll
    for (int ks = 0; ks < 2; ++ks) {     // K = 64 -> two K=32 WMMA steps
        const int kb = ks * 32;

        // A fragments (16x32 bf16): lane holds row M=lm, K-runs [8g,8g+8) and [16+8g,16+8g+8)
        v16bf aHi[2], aLo[2];
#pragma unroll
        for (int t = 0; t < 2; ++t) {
            const int m = wm0 + t * 16 + lm;
            const unsigned short* ph = &sAhi[m * RS + kb + 8 * g];
            const unsigned short* pl = &sAlo[m * RS + kb + 8 * g];
            aHi[t] = cat8(*(const v8bf*)ph, *(const v8bf*)(ph + 16));
            aLo[t] = cat8(*(const v8bf*)pl, *(const v8bf*)(pl + 16));
        }

        // B fragments (32x16 bf16): lane holds col N=lm, K-runs [8g,8g+8) and [16+8g,16+8g+8)
        v16bf bHi[4], bLo[4];
#pragma unroll
        for (int u = 0; u < 4; ++u) {
            const int n = wn0 + u * 16 + lm;
            const unsigned short* ph = &sBhi[n * RS + kb + 8 * g];
            const unsigned short* pl = &sBlo[n * RS + kb + 8 * g];
            bHi[u] = cat8(*(const v8bf*)ph, *(const v8bf*)(ph + 16));
            bLo[u] = cat8(*(const v8bf*)pl, *(const v8bf*)(pl + 16));
        }

        // 3-pass hi/lo split: A*B ~= Ahi*Bhi + Ahi*Blo + Alo*Bhi  (f32 accumulate)
#pragma unroll
        for (int t = 0; t < 2; ++t)
#pragma unroll
            for (int u = 0; u < 4; ++u) {
                acc[t][u] = __builtin_amdgcn_wmma_f32_16x16x32_bf16(
                    false, aHi[t], false, bHi[u], (short)0, acc[t][u], false, false);
                acc[t][u] = __builtin_amdgcn_wmma_f32_16x16x32_bf16(
                    false, aHi[t], false, bLo[u], (short)0, acc[t][u], false, false);
                acc[t][u] = __builtin_amdgcn_wmma_f32_16x16x32_bf16(
                    false, aLo[t], false, bHi[u], (short)0, acc[t][u], false, false);
            }
    }

    // ---------------- streaming (non-temporal) store of C tiles ----------------
    // C/D layout: VGPR j -> M = j + 8*g, N = lm
    float* outB = out + (size_t)batch * HW * HW;
    const size_t gm0 = (size_t)tM * BT + wm0 + g * 8;
    const size_t gn0 = (size_t)tN * BT + wn0 + lm;
#pragma unroll
    for (int t = 0; t < 2; ++t)
#pragma unroll
        for (int u = 0; u < 4; ++u) {
            float* p = outB + (gm0 + (size_t)t * 16) * (size_t)HW + gn0 + (size_t)u * 16;
#pragma unroll
            for (int j = 0; j < 8; ++j)
                __builtin_nontemporal_store(acc[t][u][j], p + (size_t)j * HW);
        }
}

extern "C" void kernel_launch(void* const* d_in, const int* in_sizes, int n_in,
                              void* d_out, int out_size, void* d_ws, size_t ws_size,
                              hipStream_t stream) {
    (void)in_sizes; (void)n_in; (void)out_size; (void)d_ws; (void)ws_size;
    const float* A = (const float*)d_in[0];
    const float* B = (const float*)d_in[1];
    float* out = (float*)d_out;
    dim3 grid(HW / BT, HW / BT, NB);   // (18, 18, 8)
    dim3 block(256);                    // 8 wave32 waves
    hipLaunchKernelGGL(corr_wmma_bf16split_kernel, grid, block, 0, stream, A, B, out);
}
